// TransformerCuda_22497038697061
// MI455X (gfx1250) — compile-verified, matrix operand order checked by reading, not run on hardware
//
#include <hip/hip_runtime.h>
#include <math.h>

// ---------- native vector types ----------
typedef __bf16        v16bf  __attribute__((ext_vector_type(16)));
typedef float         v8f    __attribute__((ext_vector_type(8)));
typedef float         f32x4  __attribute__((ext_vector_type(4)));
typedef unsigned int  u32x4  __attribute__((ext_vector_type(4)));

union Frag  { u32x4 q[2]; v16bf v; };
union Pack8 { u32x4 v; unsigned short s[8]; };

__device__ __forceinline__ unsigned short f2bf(float f) {
  union { float f; unsigned int u; } c; c.f = f;
  unsigned int u = c.u;
  u += 0x7FFFu + ((u >> 16) & 1u);       // round-to-nearest-even bf16
  return (unsigned short)(u >> 16);
}

// async global->LDS b128 copy (CDNA5 TDM-adjacent path, ASYNCcnt-tracked).
// ldsaddr: low 32 bits of a generic pointer into __shared__ (ISA 10.2: LDS offset).
__device__ __forceinline__ void async_copy_b128(unsigned ldsaddr, const void* gaddr) {
  asm volatile("global_load_async_to_lds_b128 %0, %1, off"
               :: "v"(ldsaddr), "v"(gaddr) : "memory");
}
__device__ __forceinline__ void async_wait0() {
  asm volatile("s_wait_asynccnt 0" ::: "memory");
}

// =====================================================================================
// f32 -> bf16 bulk convert (one shot per tensor; n must be a multiple of 2048)
// =====================================================================================
__global__ __launch_bounds__(256)
void f32_to_bf16_kern(const float* __restrict__ in, unsigned short* __restrict__ out)
{
  size_t base = ((size_t)blockIdx.x * 256 + threadIdx.x) * 8;
  f32x4 a = *(const f32x4*)&in[base];
  f32x4 b = *(const f32x4*)&in[base + 4];
  u32x4 o;
  o.x = (unsigned)f2bf(a.x) | ((unsigned)f2bf(a.y) << 16);
  o.y = (unsigned)f2bf(a.z) | ((unsigned)f2bf(a.w) << 16);
  o.z = (unsigned)f2bf(b.x) | ((unsigned)f2bf(b.y) << 16);
  o.w = (unsigned)f2bf(b.z) | ((unsigned)f2bf(b.w) << 16);
  *(u32x4*)&out[base] = o;
}

// =====================================================================================
// Tiled GEMM on bf16 operands: C(f32) = epi(alpha * A @ B), optional bf16 C copy.
//   A: [M,K] bf16 row-major.  B: transB ? [N,K] : [K,N] bf16 row-major.
//   BM=BN=128, BK=32, double-buffered LDS, async global->LDS staging.
//   256 threads = 8 waves in a 4x2 grid; wave = 32 rows x 64 cols
//   -> 2 A-frags + 4 B-frags + 8 v_wmma_f32_16x16x32_bf16 per k-step.
// =====================================================================================
__global__ __launch_bounds__(256)
void gemm_bf16_wmma(const unsigned short* __restrict__ A, int lda, long long sAb,
                    const unsigned short* __restrict__ B, int ldb, long long sBb, int transB,
                    float* __restrict__ C, int ldc, long long sCb,
                    unsigned short* __restrict__ Cbf,
                    const float* __restrict__ Add, long long sDb,
                    int K, float alpha, int doRelu)
{
  __shared__ unsigned short As[2][128 * 32];   // [row][k]
  __shared__ unsigned short Bs[2][128 * 32];   // [col][k] (K-contiguous per column)

  A += (size_t)blockIdx.z * sAb;
  B += (size_t)blockIdx.z * sBb;
  C += (size_t)blockIdx.z * sCb;
  if (Cbf) Cbf += (size_t)blockIdx.z * sCb;
  if (Add) Add += (size_t)blockIdx.z * sDb;

  const int rowBase = blockIdx.y * 128;
  const int colBase = blockIdx.x * 128;

  const int tid  = threadIdx.x;
  const int lane = tid & 31;
  const int wave = tid >> 5;
  const int wr   = wave & 3;        // row group: 32 rows
  const int wc   = wave >> 2;       // col group: 64 cols
  const int l15  = lane & 15;
  const int kb   = (lane >> 4) * 8;
  const int nk   = K >> 5;

  v8f acc[2][4];
#pragma unroll
  for (int i = 0; i < 2; ++i)
#pragma unroll
    for (int c = 0; c < 4; ++c)
#pragma unroll
      for (int r = 0; r < 8; ++r) acc[i][c][r] = 0.0f;

  auto stage = [&](int kt, int buf) {
    const int kbase = kt * 32;
    // A tile: straight copy, async global->LDS
#pragma unroll
    for (int i = 0; i < 2; ++i) {
      int e  = tid + i * 256;           // 0..511
      int r  = e >> 2;                  // 0..127
      int kq = (e & 3) << 3;            // 0,8,16,24
      async_copy_b128((unsigned)(size_t)&As[buf][r * 32 + kq],
                      &A[(size_t)(rowBase + r) * lda + kbase + kq]);
    }
    if (transB) {                       // B is [N,K]: straight copy, async
#pragma unroll
      for (int i = 0; i < 2; ++i) {
        int e  = tid + i * 256;
        int n  = e >> 2;
        int kq = (e & 3) << 3;
        async_copy_b128((unsigned)(size_t)&Bs[buf][n * 32 + kq],
                        &B[(size_t)(colBase + n) * ldb + kbase + kq]);
      }
    } else {                            // B is [K,N]: transpose while staging (VGPR path)
#pragma unroll
      for (int i = 0; i < 2; ++i) {
        int e  = tid + i * 256;
        int k  = e >> 4;                // 0..31
        int nq = (e & 15) << 3;         // 0,8,..,120
        Pack8 u;
        u.v = *(const u32x4*)&B[(size_t)(kbase + k) * ldb + colBase + nq];
#pragma unroll
        for (int j = 0; j < 8; ++j) Bs[buf][(nq + j) * 32 + k] = u.s[j];
      }
    }
  };

  stage(0, 0);

  for (int kt = 0; kt < nk; ++kt) {
    const int buf = kt & 1;
    async_wait0();
    __syncthreads();                    // tile kt staged; tile kt-1 reads done
    if (kt + 1 < nk) stage(kt + 1, buf ^ 1);

    Frag fa0, fa1;
    {
      const u32x4* ap = (const u32x4*)&As[buf][(wr * 32 + l15) * 32 + kb];
      fa0.q[0] = ap[0]; fa0.q[1] = ap[2];
    }
    {
      const u32x4* ap = (const u32x4*)&As[buf][(wr * 32 + 16 + l15) * 32 + kb];
      fa1.q[0] = ap[0]; fa1.q[1] = ap[2];
    }
#pragma unroll
    for (int c = 0; c < 4; ++c) {
      Frag fb;
      const u32x4* bp = (const u32x4*)&Bs[buf][(wc * 64 + c * 16 + l15) * 32 + kb];
      fb.q[0] = bp[0]; fb.q[1] = bp[2];
      acc[0][c] = __builtin_amdgcn_wmma_f32_16x16x32_bf16(
          false, fa0.v, false, fb.v, (short)0, acc[0][c], false, false);
      acc[1][c] = __builtin_amdgcn_wmma_f32_16x16x32_bf16(
          false, fa1.v, false, fb.v, (short)0, acc[1][c], false, false);
    }
  }

  // ---- epilogue: scale, optional residual, optional relu, f32 + optional bf16 out ----
#pragma unroll
  for (int i = 0; i < 2; ++i) {
#pragma unroll
    for (int c = 0; c < 4; ++c) {
#pragma unroll
      for (int r = 0; r < 8; ++r) {
        int m = r + (lane >> 4) * 8;
        size_t row = (size_t)(rowBase + wr * 32 + i * 16 + m);
        size_t col = (size_t)(colBase + wc * 64 + c * 16 + l15);
        float v = acc[i][c][r] * alpha;
        if (Add)    v += Add[row * ldc + col];
        if (doRelu) v = fmaxf(v, 0.0f);
        C[row * ldc + col] = v;
        if (Cbf) Cbf[row * ldc + col] = f2bf(v);
      }
    }
  }
}

// =====================================================================================
// Row softmax over f32 rows of length 2048; writes bf16 weights IN PLACE over the
// first half of each f32 row (all reads complete before the reduction barriers).
// =====================================================================================
__global__ __launch_bounds__(256)
void softmax2048_bf16(float* __restrict__ g)
{
  float* p = g + (size_t)blockIdx.x * 2048;
  unsigned short* q = (unsigned short*)p;     // bf16 row alias, 2048 entries
  const int t = threadIdx.x;
  __shared__ float red[256];

  float v[8];
#pragma unroll
  for (int i = 0; i < 8; ++i) v[i] = p[t + i * 256];

  float m = v[0];
#pragma unroll
  for (int i = 1; i < 8; ++i) m = fmaxf(m, v[i]);
  red[t] = m; __syncthreads();
  for (int off = 128; off > 0; off >>= 1) {
    if (t < off) red[t] = fmaxf(red[t], red[t + off]);
    __syncthreads();
  }
  m = red[0]; __syncthreads();

  float s = 0.0f;
#pragma unroll
  for (int i = 0; i < 8; ++i) { v[i] = __expf(v[i] - m); s += v[i]; }
  red[t] = s; __syncthreads();
  for (int off = 128; off > 0; off >>= 1) {
    if (t < off) red[t] += red[t + off];
    __syncthreads();
  }
  const float inv = 1.0f / red[0];
#pragma unroll
  for (int i = 0; i < 8; ++i) q[t + i * 256] = f2bf(v[i] * inv);
}

// =====================================================================================
// Layernorm ((x-mean)/std, ddof=1, no eps), row 1024; f32 out + optional bf16 out.
// =====================================================================================
__global__ __launch_bounds__(256)
void layernorm1024(const float* __restrict__ in, float* __restrict__ out,
                   unsigned short* __restrict__ outbf)
{
  const float* p = in  + (size_t)blockIdx.x * 1024;
  float*       o = out + (size_t)blockIdx.x * 1024;
  const int t = threadIdx.x;
  __shared__ float red[256];

  float v[4];
  float s = 0.0f;
#pragma unroll
  for (int i = 0; i < 4; ++i) { v[i] = p[t + i * 256]; s += v[i]; }
  red[t] = s; __syncthreads();
  for (int off = 128; off > 0; off >>= 1) {
    if (t < off) red[t] += red[t + off];
    __syncthreads();
  }
  const float mean = red[0] * (1.0f / 1024.0f);
  __syncthreads();

  float ss = 0.0f;
#pragma unroll
  for (int i = 0; i < 4; ++i) { float d = v[i] - mean; ss += d * d; }
  red[t] = ss; __syncthreads();
  for (int off = 128; off > 0; off >>= 1) {
    if (t < off) red[t] += red[t + off];
    __syncthreads();
  }
  const float rstd = rsqrtf(red[0] * (1.0f / 1023.0f));   // ddof = 1
#pragma unroll
  for (int i = 0; i < 4; ++i) {
    float r = (v[i] - mean) * rstd;
    o[t + i * 256] = r;
    if (outbf) outbf[(size_t)blockIdx.x * 1024 + t + i * 256] = f2bf(r);
  }
}

// =====================================================================================
// out[b] = mean_s(x[b,s,:]) . readout / sqrt(1024)
// =====================================================================================
__global__ __launch_bounds__(256)
void pool_readout(const float* __restrict__ xf, const float* __restrict__ r,
                  float* __restrict__ out)
{
  const int b = blockIdx.x;
  const int t = threadIdx.x;
  const float* xb = xf + (size_t)b * 2048 * 1024;
  float acc = 0.0f;
  for (int s = 0; s < 2048; ++s) {
    const float* row = xb + (size_t)s * 1024;
#pragma unroll
    for (int i = 0; i < 4; ++i) {
      int d = t + i * 256;
      acc += row[d] * r[d];
    }
  }
  __shared__ float red[256];
  red[t] = acc; __syncthreads();
  for (int off = 128; off > 0; off >>= 1) {
    if (t < off) red[t] += red[t + off];
    __syncthreads();
  }
  if (t == 0) out[b] = red[0] * (1.0f / (2048.0f * 32.0f));
}

// =====================================================================================
extern "C" void kernel_launch(void* const* d_in, const int* in_sizes, int n_in,
                              void* d_out, int out_size, void* d_ws, size_t ws_size,
                              hipStream_t stream)
{
  (void)in_sizes; (void)n_in; (void)out_size; (void)ws_size;
  const float* seq = (const float*)d_in[0];   // [8,2048,768]
  const float* emb = (const float*)d_in[1];   // [768,1024]
  const float* W1  = (const float*)d_in[2];   // [1024,1024]
  const float* W2  = (const float*)d_in[3];   // [1024,1024]
  const float* ro  = (const float*)d_in[4];   // [1024]
  float* outp = (float*)d_out;                // [8]

  char* ws = (char*)d_ws;
  float*          Xf   = (float*)(ws + 0);                  //  67 MB
  float*          Gf   = (float*)(ws + 67108864);           // 134 MB gram; also hosts
                                                            //   seqbf/embbf early, MLP1 f32 dump late
  float*          Tf   = (float*)(ws + 201326592);          //  67 MB
  float*          Pf   = (float*)(ws + 268435456);          //  67 MB
  unsigned short* Xbf  = (unsigned short*)(ws + 335544320); //  33.5 MB
  unsigned short* Pbf  = (unsigned short*)(ws + 369098752); //  33.5 MB
  unsigned short* Hbf  = (unsigned short*)(ws + 402653184); //  33.5 MB
  unsigned short* W1bf = (unsigned short*)(ws + 436207616); //   2 MB
  unsigned short* W2bf = (unsigned short*)(ws + 438304768); //   2 MB
  // transient bf16 inputs live inside the (not-yet-written) gram region:
  unsigned short* Seqbf = (unsigned short*)((char*)Gf);                // 25.2 MB
  unsigned short* Embbf = (unsigned short*)((char*)Gf + 25165824);     //  1.5 MB
  unsigned short* Wbf   = (unsigned short*)Gf;   // softmax writes bf16 weights in place,
                                                 // row stride 4096 bf16 (= 2048 f32)

  const long long sX = 2048LL * 1024;
  const long long sG = 2048LL * 2048;

  // 0) one-shot input conversions
  f32_to_bf16_kern<<<6144, 256, 0, stream>>>(seq, Seqbf);
  f32_to_bf16_kern<<< 384, 256, 0, stream>>>(emb, Embbf);
  f32_to_bf16_kern<<< 512, 256, 0, stream>>>(W1,  W1bf);
  f32_to_bf16_kern<<< 512, 256, 0, stream>>>(W2,  W2bf);

  // 1) X = seq @ emb / sqrt(768)   (f32 + bf16 outputs)   M=16384 K=768 N=1024
  gemm_bf16_wmma<<<dim3(8, 128, 1), 256, 0, stream>>>(
      Seqbf, 768, 0, Embbf, 1024, 0, /*transB=*/0,
      Xf, 1024, 0, Xbf, nullptr, 0, 768, 1.0f / sqrtf(768.0f), 0);

  // 2) G = X @ X^T / 1024 (per batch)   M=2048 K=1024 N=2048  (overwrites Seqbf region)
  gemm_bf16_wmma<<<dim3(16, 16, 8), 256, 0, stream>>>(
      Xbf, 1024, sX, Xbf, 1024, sX, /*transB=*/1,
      Gf, 2048, sG, nullptr, nullptr, 0, 1024, 1.0f / 1024.0f, 0);

  // 3) softmax rows of G -> bf16 weights in place (row stride 4096 bf16)
  softmax2048_bf16<<<16384, 256, 0, stream>>>(Gf);

  // 4) T = W @ X + X (per batch)   M=2048 K=2048 N=1024
  gemm_bf16_wmma<<<dim3(8, 16, 8), 256, 0, stream>>>(
      Wbf, 4096, 2048LL * 4096, Xbf, 1024, sX, /*transB=*/0,
      Tf, 1024, sX, nullptr, Xf, sX, 2048, 1.0f, 0);

  // 5) P = layernorm(T)  (f32 + bf16)
  layernorm1024<<<16384, 256, 0, stream>>>(Tf, Pf, Pbf);

  // 6) H = relu(P @ W1 / 32) -> bf16 (f32 dump into free gram region)
  gemm_bf16_wmma<<<dim3(8, 128, 1), 256, 0, stream>>>(
      Pbf, 1024, 0, W1bf, 1024, 0, /*transB=*/0,
      Gf, 1024, 0, Hbf, nullptr, 0, 1024, 1.0f / 32.0f, 1);

  // 7) T = H @ W2 / 32 + P
  gemm_bf16_wmma<<<dim3(8, 128, 1), 256, 0, stream>>>(
      Hbf, 1024, 0, W2bf, 1024, 0, /*transB=*/0,
      Tf, 1024, 0, nullptr, Pf, 0, 1024, 1.0f / 32.0f, 0);

  // 8) X = layernorm(T)   (final activations, f32 only)
  layernorm1024<<<16384, 256, 0, stream>>>(Tf, Xf, nullptr);

  // 9) out[b] = mean_s(X) . readout / 32
  pool_readout<<<8, 256, 0, stream>>>(Xf, ro, outp);
}
